// Attention_3891240370664
// MI455X (gfx1250) — matrix operand order repaired and checked by reading, not compiled, not run on hardware
//
#include <hip/hip_runtime.h>

// ---------------------------------------------------------------------------
// Fused attention for B=2,H=16,S=2048,D=128 (fp32 in / fp32 out) on gfx1250.
// Strategy: bf16 WMMA (v_wmma_f32_16x16x32_bf16) for both matmuls, flash-style
// two-pass softmax so scores never hit HBM; p_attn is written exactly once.
// Workspace: Q,K as bf16 (row major) + V as bf16 transposed (d-major),
// 3 * 16.78 MB = ~50 MB in d_ws; fits comfortably in the 192 MB L2 so K/V
// re-reads across query tiles stay on-chip.
// ---------------------------------------------------------------------------

#define Bn 2
#define Hn 16
#define Sn 2048
#define Dn 128
#define BHn (Bn * Hn)

typedef __bf16 v16bf __attribute__((ext_vector_type(16)));
typedef float  v8f   __attribute__((ext_vector_type(8)));

union Frag {
  v16bf v;
  uint4 q[2];   // 32 bytes = 16 bf16
};

__device__ __forceinline__ unsigned short f2bf(float x) {
  unsigned int u = __float_as_uint(x);
  u += 0x7FFFu + ((u >> 16) & 1u);       // round-to-nearest-even
  return (unsigned short)(u >> 16);
}

// 16 contiguous bf16 starting at p (32B, 16B-aligned)
__device__ __forceinline__ Frag load_contig32(const unsigned short* p) {
  Frag f;
  f.q[0] = *(const uint4*)(p);
  f.q[1] = *(const uint4*)(p + 8);
  return f;
}

// ---------------------------------------------------------------------------
// Prologue: fp32 -> bf16 conversion (layout preserved), vectorized x4.
// ---------------------------------------------------------------------------
__global__ void cvt_f32_bf16(const float* __restrict__ in,
                             unsigned short* __restrict__ out, int n) {
  int i = (blockIdx.x * blockDim.x + threadIdx.x) * 4;
  if (i < n) {
    float4 v = *(const float4*)(in + i);
    ushort4 o;
    o.x = f2bf(v.x); o.y = f2bf(v.y); o.z = f2bf(v.z); o.w = f2bf(v.w);
    *(ushort4*)(out + i) = o;
  }
}

// ---------------------------------------------------------------------------
// Prologue: V (b,h,n,d) fp32 -> Vt (b,h,d,n) bf16, tiled 32x32 transpose.
// Makes the PV B-fragment a contiguous 32B load per lane.
// ---------------------------------------------------------------------------
__global__ void transpose_v_bf16(const float* __restrict__ V,
                                 unsigned short* __restrict__ Vt) {
  __shared__ float tile[32][33];
  const int bh = blockIdx.z;
  const float* Vb = V + (size_t)bh * Sn * Dn;
  unsigned short* Tb = Vt + (size_t)bh * Dn * Sn;
  const int n0 = blockIdx.x * 32;
  const int d0 = blockIdx.y * 32;
#pragma unroll
  for (int i = 0; i < 4; ++i) {
    int n = threadIdx.y + i * 8;
    tile[n][threadIdx.x] = Vb[(size_t)(n0 + n) * Dn + d0 + threadIdx.x];
  }
  __syncthreads();
#pragma unroll
  for (int i = 0; i < 4; ++i) {
    int d = threadIdx.y + i * 8;
    Tb[(size_t)(d0 + d) * Sn + n0 + threadIdx.x] = f2bf(tile[threadIdx.x][d]);
  }
}

// ---------------------------------------------------------------------------
// Fused attention kernel. 128 threads = 4 wave32, each wave owns 16 q-rows.
// Pass A: online softmax stats (exp2 domain) over all keys via WMMA.
// Pass B: recompute scores, write normalized p_attn once, bounce p through a
// per-wave LDS tile (C/D layout -> A layout) and accumulate P*V via WMMA.
// ---------------------------------------------------------------------------
__global__ __launch_bounds__(128) void attn_fused(
    const unsigned short* __restrict__ Qb16,
    const unsigned short* __restrict__ Kb16,
    const unsigned short* __restrict__ Vtb16,
    float* __restrict__ out_pv,
    float* __restrict__ out_pa) {
  const int wave = threadIdx.x >> 5;
  const int lane = threadIdx.x & 31;
  const int half = lane >> 4;   // which 16-lane half of the wave
  const int l16  = lane & 15;
  const int bh   = blockIdx.y;
  const int qbase = blockIdx.x * 64 + wave * 16;

  const unsigned short* Q  = Qb16  + (size_t)bh * Sn * Dn;
  const unsigned short* K  = Kb16  + (size_t)bh * Sn * Dn;
  const unsigned short* Vt = Vtb16 + (size_t)bh * Dn * Sn;
  float* pa = out_pa + (size_t)bh * Sn * Sn;
  float* pv = out_pv + (size_t)bh * Sn * Dn;

  // per-wave p staging tile, row stride padded to 40 bf16 (80B) for banks
  __shared__ __align__(16) unsigned short plds[4][16][40];

  // (1/sqrt(128)) * log2(e): softmax computed entirely in exp2 domain
  const float SCALE = 0.08838834764831845f * 1.4426950408889634f;

  // --- load Q A-fragments (16x128 strip -> 4 fragments of 16x32) ---
  // A layout: lane l16 holds row l16; half selects contiguous k-cols c0..c0+15
  Frag a[4];
  {
    const unsigned short* qrow = Q + (size_t)(qbase + l16) * Dn;
#pragma unroll
    for (int kf = 0; kf < 4; ++kf) {
      const int c0 = kf * 32 + half * 16;
      a[kf].q[0] = *(const uint4*)(qrow + c0);
      a[kf].q[1] = *(const uint4*)(qrow + c0 + 8);
    }
  }

  // --- Pass A: online max/sum, per-lane partials over this lane's columns ---
  float mrow[8], srow[8];
#pragma unroll
  for (int r = 0; r < 8; ++r) { mrow[r] = -__builtin_inff(); srow[r] = 0.0f; }

  for (int n0 = 0; n0 < Sn; n0 += 16) {
    const unsigned short* krow = K + (size_t)(n0 + l16) * Dn;
    __builtin_prefetch(krow + 16 * Dn, 0, 1);   // global_prefetch_b8 next tile
    v8f acc = {};
#pragma unroll
    for (int kf = 0; kf < 4; ++kf) {
      // B layout: lane = key column (K row), 16 contiguous d-halves per lane
      Frag b = load_contig32(krow + kf * 32 + half * 16);
      acc = __builtin_amdgcn_wmma_f32_16x16x32_bf16(
          false, a[kf].v, false, b.v, (short)0, acc, false, false);
    }
#pragma unroll
    for (int r = 0; r < 8; ++r) {
      float v  = acc[r] * SCALE;          // row (r + 8*half), col (n0 + l16)
      float mo = mrow[r];
      float mn = fmaxf(mo, v);
      srow[r] = srow[r] * __builtin_amdgcn_exp2f(mo - mn)
                        + __builtin_amdgcn_exp2f(v - mn);
      mrow[r] = mn;
    }
  }

  // merge the 16 lanes that share each row (xor masks < 16 keep half fixed)
  float inv[8];
#pragma unroll
  for (int r = 0; r < 8; ++r) {
    float mx = mrow[r], sm = srow[r];
#pragma unroll
    for (int d = 1; d < 16; d <<= 1) {
      float mo = __shfl_xor(mx, d, 32);
      float so = __shfl_xor(sm, d, 32);
      float mn = fmaxf(mx, mo);
      sm = sm * __builtin_amdgcn_exp2f(mx - mn)
         + so * __builtin_amdgcn_exp2f(mo - mn);
      mx = mn;
    }
    mrow[r] = mx;
    inv[r] = 1.0f / sm;
  }

  // --- Pass B: recompute scores, emit p_attn, accumulate P*V ---
  v8f o[8];
#pragma unroll
  for (int dt = 0; dt < 8; ++dt) o[dt] = (v8f){};

  for (int n0 = 0; n0 < Sn; n0 += 32) {
#pragma unroll
    for (int t = 0; t < 2; ++t) {
      const int nb = n0 + t * 16;
      const unsigned short* krow = K + (size_t)(nb + l16) * Dn;
      v8f acc = {};
#pragma unroll
      for (int kf = 0; kf < 4; ++kf) {
        Frag b = load_contig32(krow + kf * 32 + half * 16);
        acc = __builtin_amdgcn_wmma_f32_16x16x32_bf16(
            false, a[kf].v, false, b.v, (short)0, acc, false, false);
      }
#pragma unroll
      for (int r = 0; r < 8; ++r) {
        const int lrow = r + 8 * half;
        float p = __builtin_amdgcn_exp2f(acc[r] * SCALE - mrow[r]) * inv[r];
        pa[(size_t)(qbase + lrow) * Sn + nb + l16] = p;       // normalized p_attn
        plds[wave][lrow][t * 16 + l16] = f2bf(p);             // stage for PV
      }
    }
    // Re-read the 16x32 p tile in A-fragment layout (wave-private LDS; DS ops
    // from one wave are in-order, compiler inserts s_wait_dscnt).
    Frag pf;
    {
      const unsigned short* prow = &plds[wave][l16][half * 16];
      pf.q[0] = *(const uint4*)(prow);
      pf.q[1] = *(const uint4*)(prow + 8);
    }
#pragma unroll
    for (int dt = 0; dt < 8; ++dt) {
      // V B-fragment from transposed V: lane = d column, contiguous along n
      const unsigned short* vrow =
          Vt + (size_t)(dt * 16 + l16) * Sn + n0 + half * 16;
      Frag vb = load_contig32(vrow);
      o[dt] = __builtin_amdgcn_wmma_f32_16x16x32_bf16(
          false, pf.v, false, vb.v, (short)0, o[dt], false, false);
    }
  }

  // --- epilogue: write p_val ---
#pragma unroll
  for (int dt = 0; dt < 8; ++dt)
#pragma unroll
    for (int r = 0; r < 8; ++r)
      pv[(size_t)(qbase + r + 8 * half) * Dn + dt * 16 + l16] = o[dt][r];
}

// ---------------------------------------------------------------------------
extern "C" void kernel_launch(void* const* d_in, const int* in_sizes, int n_in,
                              void* d_out, int out_size, void* d_ws, size_t ws_size,
                              hipStream_t stream) {
  const float* Q = (const float*)d_in[0];
  const float* K = (const float*)d_in[1];
  const float* V = (const float*)d_in[2];
  // d_in[3] ("m") is falsy in the reference -> ignored.

  const size_t NE = (size_t)Bn * Hn * Sn * Dn;   // 8,388,608 elements
  float* out = (float*)d_out;
  float* pv = out;            // p_val: B*H*S*D floats, first in tuple order
  float* pa = out + NE;       // p_attn: B*H*S*S floats

  // workspace: [Q bf16 | K bf16 | Vt bf16] = 3 * 16.78 MB ~= 50 MB
  unsigned short* wsQ  = (unsigned short*)d_ws;
  unsigned short* wsK  = wsQ + NE;
  unsigned short* wsVt = wsK + NE;

  const int nthr = 256;
  const int nblk = (int)(NE / 4 / nthr);
  cvt_f32_bf16<<<nblk, nthr, 0, stream>>>(Q, wsQ, (int)NE);
  cvt_f32_bf16<<<nblk, nthr, 0, stream>>>(K, wsK, (int)NE);

  dim3 tgrid(Sn / 32, Dn / 32, BHn);
  dim3 tblk(32, 8);
  transpose_v_bf16<<<tgrid, tblk, 0, stream>>>(V, wsVt);

  dim3 agrid(Sn / 64, BHn);   // 64 q-rows per 128-thread (4-wave) block
  attn_fused<<<agrid, 128, 0, stream>>>(wsQ, wsK, wsVt, pv, pa);
}
